// VQ_1365799600221
// MI455X (gfx1250) — matrix operand order repaired and checked by reading, not compiled, not run on hardware
//
#include <hip/hip_runtime.h>

typedef __attribute__((ext_vector_type(16))) __bf16       v16bf;
typedef __attribute__((ext_vector_type(8)))  float        v8f;
typedef __attribute__((ext_vector_type(4)))  float        v4f;
typedef __attribute__((ext_vector_type(4)))  unsigned int v4u;

#define N_ROWS   131072      // 32*64*64
#define DIM      64
#define K_CODES  512
#define NBLOCKS  256         // N_ROWS / 512 rows per block
#define BIAS     16.0f       // makes every score positive -> u32-orderable

// workspace layout (bytes)
#define PART_OFF 0                      // 256 f32 partial loss sums
#define NORM_OFF 1024                   // 512 f32 biased codebook column norms
#define CT_OFF   4096                   // 512x64 f32 transposed codebook (gather table)
#define BF_OFF   (4096 + 131072)        // 32768 bf16 pre-swizzled A fragments (64 KB)

__device__ __forceinline__ unsigned umin2(unsigned a, unsigned b) { return a < b ? a : b; }

// ---------------------------------------------------------------------------
// Prep: biased column norms (||c||^2 + BIAS), f32 transposed codebook for the
// output gather, and bf16 codebook PRE-SCALED BY -2 swizzled into the CDNA5
// 16-bit A-MATRIX lane layout (codes are the M dimension now):
//   code n = t*16+nn, inner d: dd = d&31, chunk c = d>>5
//   lane = ((dd>>3)&1)*16 + nn,  j = ((dd>>4)&1)*8 + (dd&7)
//   -> bfr[(t*2+c)*512 + lane*16 + j]
// so each lane reads its 16 contiguous bf16 (32B) per (tile, chunk).
// ---------------------------------------------------------------------------
__global__ __launch_bounds__(256) void vq_prep(const float* __restrict__ cb,
                                               __bf16* __restrict__ bfr,
                                               float* __restrict__ norms,
                                               float* __restrict__ ct)
{
    int n  = blockIdx.x * 256 + threadIdx.x;   // 0..511
    int t  = n >> 4, nn = n & 15;
    float s = 0.f;
    for (int d = 0; d < DIM; ++d) {
        float v = cb[d * K_CODES + n];
        s += v * v;
        ct[n * DIM + d] = v;                    // exact values for the gather
        int c = d >> 5, dd = d & 31;
        int lane = ((dd >> 3) & 1) * 16 + nn;
        int j    = ((dd >> 4) & 1) * 8 + (dd & 7);
        bfr[(t * 2 + c) * 512 + lane * 16 + j] = (__bf16)(-2.f * v);
    }
    norms[n] = s + BIAS;
}

// ---------------------------------------------------------------------------
// Main fused kernel. Codes = M (A from LDS), rows = N (B in registers).
// D layout: VGPR r, lanes 0-15 -> (code t*16+r, row ln);
//           lanes 16-31 -> (code t*16+8+r, row ln).
// Each lane owns 8 codes of ONE row -> argmin via packed u32 keys:
//   key = (float_bits(score) & ~0xFF) | (t*8 + r), score > 0 guaranteed.
// Block = 256 threads = 8 waves; wave handles 64 rows (4 x 16-row B tiles).
// ---------------------------------------------------------------------------
__global__ __launch_bounds__(256) void vq_main(const float* __restrict__ x,
                                               const v4u* __restrict__ bfr,
                                               const float* __restrict__ norms,
                                               const float* __restrict__ ct,
                                               float* __restrict__ out,
                                               float* __restrict__ partial)
{
    __shared__ v4u   Alds[4096];      // 64 KB: all 512 codes' A fragments
    __shared__ float nlds[512];       // biased code norms
    __shared__ int   idx_lds[512];    // argmin per row of this block
    __shared__ float wsum[8];

    const int tid = threadIdx.x;

    for (int i = tid; i < 4096; i += 256) Alds[i] = bfr[i];
    for (int i = tid; i < 512;  i += 256) nlds[i] = norms[i];
    __syncthreads();

    const int wave = tid >> 5, lane = tid & 31;
    const int g = lane >> 4, ln = lane & 15;   // code-half / row-in-tile
    const long wbase = ((long)blockIdx.x * 8 + wave) * 64;

    // B fragments: rows of x, bf16, per the 16-bit B-matrix layout:
    // lane holds row N=ln, K = g*8 + {0..7} and 16 + g*8 + {0..7} of chunk c
    v16bf bfrg[4][2];
#pragma unroll
    for (int rt = 0; rt < 4; ++rt) {
#pragma unroll
        for (int c = 0; c < 2; ++c) {
            const float* rp = x + (wbase + rt * 16 + ln) * DIM + c * 32 + g * 8;
            v4f f0 = *(const v4f*)(rp);
            v4f f1 = *(const v4f*)(rp + 4);
            v4f f2 = *(const v4f*)(rp + 16);
            v4f f3 = *(const v4f*)(rp + 20);
            v16bf b;
#pragma unroll
            for (int j = 0; j < 4; ++j) {
                b[j]      = (__bf16)f0[j];
                b[4 + j]  = (__bf16)f1[j];
                b[8 + j]  = (__bf16)f2[j];
                b[12 + j] = (__bf16)f3[j];
            }
            bfrg[rt][c] = b;
        }
    }

    unsigned run[4];
#pragma unroll
    for (int rt = 0; rt < 4; ++rt) run[rt] = 0xFFFFFFFFu;

    union AU { v4u u[2]; v16bf v; };
    union NU { v4f f[2]; v8f v; };

    for (int t = 0; t < 32; ++t) {
        AU a0, a1;
        a0.u[0] = Alds[(t * 2) * 64 + lane * 2];
        a0.u[1] = Alds[(t * 2) * 64 + lane * 2 + 1];
        a1.u[0] = Alds[(t * 2 + 1) * 64 + lane * 2];
        a1.u[1] = Alds[(t * 2 + 1) * 64 + lane * 2 + 1];

        NU bn;   // biased norms of this lane's 8 codes (C init), shared by all rt
        const v4f* np = (const v4f*)&nlds[t * 16 + g * 8];
        bn.f[0] = np[0];
        bn.f[1] = np[1];

        unsigned pk[8];
        const unsigned tb = (unsigned)(t << 3);
#pragma unroll
        for (int r = 0; r < 8; ++r) pk[r] = tb + (unsigned)r;

#pragma unroll
        for (int rt = 0; rt < 4; ++rt) {
            v8f acc = __builtin_amdgcn_wmma_f32_16x16x32_bf16(
                false, a0.v, false, bfrg[rt][0], (short)0, bn.v, false, false);
            acc = __builtin_amdgcn_wmma_f32_16x16x32_bf16(
                false, a1.v, false, bfrg[rt][1], (short)0, acc, false, false);
            unsigned kk[8];
#pragma unroll
            for (int r = 0; r < 8; ++r)
                kk[r] = (__float_as_uint(acc[r]) & 0xFFFFFF00u) | pk[r];  // v_and_or_b32
            unsigned m0 = umin2(kk[0], kk[1]);
            unsigned m1 = umin2(kk[2], kk[3]);
            unsigned m2 = umin2(kk[4], kk[5]);
            unsigned m3 = umin2(kk[6], kk[7]);
            run[rt] = umin2(run[rt], umin2(umin2(m0, m1), umin2(m2, m3)));
        }
    }

    // Merge the two code-halves of each row (lane ln <-> lane ln+16), decode.
#pragma unroll
    for (int rt = 0; rt < 4; ++rt) {
        unsigned k  = run[rt];
        unsigned ok = (unsigned)__shfl_xor((int)k, 16, 32);
        unsigned myloc = k & 255u, oloc = ok & 255u;
        int mycode = (int)((myloc >> 3) << 4) + g * 8 + (int)(myloc & 7u);
        int ocode  = (int)((oloc  >> 3) << 4) + (1 - g) * 8 + (int)(oloc & 7u);
        unsigned mys = k & ~255u, os = ok & ~255u;
        int code = (os < mys || (os == mys && ocode < mycode)) ? ocode : mycode;
        if (g == 0) idx_lds[wave * 64 + rt * 16 + ln] = code;
    }
    __syncthreads();

    // Output: out = gathered codebook row; accumulate (q - x)^2.
    float lsum = 0.f;
    const long brow = (long)blockIdx.x * 512;
    const int e = tid & 63, rq = tid >> 6;
    for (int rr = rq; rr < 512; rr += 4) {
        int   idx = idx_lds[rr];
        float q   = ct[idx * DIM + e];
        long  off = (brow + rr) * DIM + e;
        float xv  = x[off];
        out[off]  = q;
        float d   = q - xv;
        lsum      = fmaf(d, d, lsum);
    }
#pragma unroll
    for (int off = 1; off < 32; off <<= 1) lsum += __shfl_xor(lsum, off, 32);
    if (lane == 0) wsum[wave] = lsum;
    __syncthreads();
    if (tid == 0) {
        float s = 0.f;
        for (int i = 0; i < 8; ++i) s += wsum[i];
        partial[blockIdx.x] = s;   // deterministic per-block partial
    }
}

// Final reduction: loss = 1.25 * mean((q - x)^2)
__global__ __launch_bounds__(256) void vq_loss(const float* __restrict__ partial,
                                               float* __restrict__ lossp)
{
    __shared__ float sm[256];
    int tid = threadIdx.x;
    sm[tid] = partial[tid];
    __syncthreads();
    for (int s = 128; s > 0; s >>= 1) {
        if (tid < s) sm[tid] += sm[tid + s];
        __syncthreads();
    }
    if (tid == 0) lossp[0] = sm[0] * (1.25f / 8388608.f);
}

extern "C" void kernel_launch(void* const* d_in, const int* in_sizes, int n_in,
                              void* d_out, int out_size, void* d_ws, size_t ws_size,
                              hipStream_t stream)
{
    const float* x  = (const float*)d_in[0];   // (32,64,64,64) f32
    const float* cb = (const float*)d_in[1];   // (64,512) f32
    float* out = (float*)d_out;                // 8388608 outputs + 1 loss
    char*  ws  = (char*)d_ws;

    float*  partial = (float*)(ws + PART_OFF);
    float*  norms   = (float*)(ws + NORM_OFF);
    float*  ct      = (float*)(ws + CT_OFF);
    __bf16* bfr     = (__bf16*)(ws + BF_OFF);

    vq_prep<<<2, 256, 0, stream>>>(cb, bfr, norms, ct);
    vq_main<<<NBLOCKS, 256, 0, stream>>>(x, (const v4u*)bfr, norms, ct, out, partial);
    vq_loss<<<1, 256, 0, stream>>>(partial, out + 8388608);
}
